// GLayer_32100585570565
// MI455X (gfx1250) — matrix operand order repaired and checked by reading, not compiled, not run on hardware
//
#include <hip/hip_runtime.h>

// Problem constants (from reference)
#define N_ROWS 131072
#define DDIM   512
#define KDIM   32
#define NFEAT  528           // K*(K+1)/2
#define FPAD   544           // padded to multiple of 32 (17*32)
#define KBLKS  17            // FPAD / 32
#define GAMMA  0.01f
#define ALPHA  1.0f
#define TOLV   1e-6f

#define MTILE     32         // rows per block
#define ZS_STRIDE 516        // 512 + 4 pad (bank-conflict break), 16B-aligned rows
#define HS_STRIDE 552        // 544 + 8 pad, rows stay 16B-aligned

// workspace float/byte offsets
#define WS_XMUU_F   0        // 32 floats x_muU
#define WS_XCU_F    32       // 32 floats x_cU
#define WS_CHANGE_F 64       // 512 floats change
#define WS_BPACK_B  2304     // byte offset of bf16 Bpack[17][512][2][16]

typedef __attribute__((ext_vector_type(16))) __bf16          v16bf;
typedef __attribute__((ext_vector_type(8)))  float           v8f;
typedef __attribute__((ext_vector_type(8)))  unsigned short  u16x8;
typedef __attribute__((ext_vector_type(4)))  float           f32x4;

union ABFrag { v16bf v; u16x8 h[2]; };

__device__ __forceinline__ unsigned short f2bf(float f) {
  union { float f; unsigned u; } c; c.f = f;
  unsigned u = c.u + 0x7FFFu + ((c.u >> 16) & 1u);   // round-to-nearest-even
  return (unsigned short)(u >> 16);
}

// ---- prep 0: x_muU[k], x_cU[k] = (x_mu|x_c) @ U ------------------------------
__global__ void prep_proj(const float* __restrict__ U, const float* __restrict__ xmu,
                          const float* __restrict__ xc, float* __restrict__ wsf) {
  int t = threadIdx.x;                    // 64 threads
  if (t < 64) {
    int k = t & 31;
    const float* vec = (t < 32) ? xmu : xc;
    float acc = 0.f;
    #pragma unroll 4
    for (int d = 0; d < DDIM; ++d) acc += vec[d] * U[d * KDIM + k];
    wsf[t] = acc;                         // [0:32]=x_muU  [32:64]=x_cU
  }
}

// ---- prep 1: change[d] = x_c - x_mu - (x_cU - x_muU) @ U^T -------------------
__global__ void prep_change(const float* __restrict__ U, const float* __restrict__ xmu,
                            const float* __restrict__ xc, float* __restrict__ wsf) {
  int d = blockIdx.x * blockDim.x + threadIdx.x;
  if (d < DDIM) {
    float acc = 0.f;
    #pragma unroll
    for (int k = 0; k < KDIM; ++k)
      acc += (wsf[WS_XCU_F + k] - wsf[WS_XMUU_F + k]) * U[d * KDIM + k];
    wsf[WS_CHANGE_F + d] = xc[d] - xmu[d] - acc;
  }
}

// ---- prep 2: Bpack[kb][c][h][j] = bf16(V[c][kb*32 + h*16 + j]) (0-padded) ----
__global__ void prep_bpack(const float* __restrict__ V, unsigned short* __restrict__ bp) {
  int idx = blockIdx.x * blockDim.x + threadIdx.x;      // KBLKS*512*32 total
  if (idx < KBLKS * DDIM * 32) {
    int q  = idx & 31;                 // h*16 + j
    int c  = (idx >> 5) & (DDIM - 1);  // output column (D index)
    int kb = idx >> 14;                // k-block
    int f  = kb * 32 + q;              // feature index in V.T row dim
    float v = (f < NFEAT) ? V[(size_t)c * NFEAT + f] : 0.f;
    bp[idx] = f2bf(v);
  }
}

// ---- main fused kernel -------------------------------------------------------
__global__ __launch_bounds__(256) void glayer_main(
    const float* __restrict__ Z, const float* __restrict__ U,
    const float* __restrict__ xmu_g, const float* __restrict__ wsf,
    const unsigned short* __restrict__ Bp, float* __restrict__ out)
{
  extern __shared__ char smem[];
  float*          Zs     = (float*)(smem);                           // 32*516 f32   (66048 B)
  unsigned short* Hs     = (unsigned short*)(smem + 66048);          // 32*552 bf16  (35328 B)
  float*          ZUs    = (float*)(smem + 101376);                  // 32*32  f32   (4096 B)
  float*          xmu_s  = ZUs + MTILE * KDIM;                       // 512 f32
  float*          chg_s  = xmu_s + DDIM;                             // 512 f32
  float*          xmuU_s = chg_s + DDIM;                             // 32
  float*          xcU_s  = xmuU_s + 32;                              // 32
  float*          kern_s = xcU_s + 32;                               // 32
  unsigned char*  iu_s   = (unsigned char*)(kern_s + 32);            // 544
  unsigned char*  ju_s   = iu_s + 544;                               // 544
  float*          red_s  = (float*)(ju_s + 544);                     // 32*8 f32 partial norms

  const int tid  = threadIdx.x;
  const int row0 = blockIdx.x * MTILE;

  // ---- phase 0: constants, triu indices, Z tile -> LDS ----
  for (int i = tid; i < DDIM; i += 256) {
    xmu_s[i] = xmu_g[i];
    chg_s[i] = wsf[WS_CHANGE_F + i];
  }
  if (tid < 32) { xmuU_s[tid] = wsf[WS_XMUU_F + tid]; xcU_s[tid] = wsf[WS_XCU_F + tid]; }
  for (int p = tid; p < NFEAT; p += 256) {
    int i = 0, rem = p;
    while (rem >= (KDIM - i)) { rem -= (KDIM - i); ++i; }
    iu_s[p] = (unsigned char)i;
    ju_s[p] = (unsigned char)(i + rem);
  }

#if defined(__HIP_DEVICE_COMPILE__) && \
    __has_builtin(__builtin_amdgcn_global_load_async_to_lds_b128) && \
    __has_builtin(__builtin_amdgcn_s_wait_asynccnt)
  {
    typedef int v4i __attribute__((vector_size(16)));
    typedef __attribute__((address_space(1))) v4i* gv4p;
    typedef __attribute__((address_space(3))) v4i* lv4p;
    for (int i = tid; i < MTILE * (DDIM / 4); i += 256) {
      int r = i >> 7, c4 = i & 127;
      const float* g = Z + (size_t)(row0 + r) * DDIM + c4 * 4;
      float*       l = Zs + r * ZS_STRIDE + c4 * 4;
      __builtin_amdgcn_global_load_async_to_lds_b128((gv4p)(void*)g, (lv4p)(void*)l, 0, 0);
    }
    __builtin_amdgcn_s_wait_asynccnt(0);
  }
#else
  for (int i = tid; i < MTILE * (DDIM / 4); i += 256) {
    int r = i >> 7, c4 = i & 127;
    f32x4 v = *(const f32x4*)(Z + (size_t)(row0 + r) * DDIM + c4 * 4);
    *(f32x4*)(Zs + r * ZS_STRIDE + c4 * 4) = v;
  }
#endif
  __syncthreads();

  // ---- phase 1: ZU = Ztile @ U  (thread: 1 row x 4 k-columns) ----
  {
    int r = tid >> 3, kq = tid & 7;
    f32x4 acc = {0.f, 0.f, 0.f, 0.f};
    const float* zr = Zs + r * ZS_STRIDE;
    #pragma unroll 4
    for (int d = 0; d < DDIM; ++d) {
      float z = zr[d];
      f32x4 u4 = *(const f32x4*)(U + d * KDIM + kq * 4);
      acc += z * u4;
    }
    *(f32x4*)(ZUs + r * KDIM + kq * 4) = acc;
  }
  // parallel Z-row-norm partials: 8 threads per row, 64 columns each
  {
    int r = tid >> 3, part = tid & 7;
    const float* zr = Zs + r * ZS_STRIDE;
    float s = 0.f;
    int d0 = part * 64;
    #pragma unroll 4
    for (int d = d0; d < d0 + 64; ++d) { float df = zr[d] - xmu_s[d]; s += df * df; }
    red_s[r * 8 + part] = s;
  }
  __syncthreads();

  // ---- phase 2: reduce norms + secant root-find + kernel value (lane-per-row) ----
  if (tid < MTILE) {
    const int r = tid;
    float zn = 0.f;
    #pragma unroll
    for (int p = 0; p < 8; ++p) zn += red_s[r * 8 + p];
    float zun = 0.f;
    #pragma unroll
    for (int k = 0; k < KDIM; ++k) { float df = ZUs[r * KDIM + k] - xmuU_s[k]; zun += df * df; }
    float zperp = zn - zun;
    float eg  = ALPHA * __expf(-GAMMA * zun);
    float xm2 = zperp * 0.999f, xm1 = zperp;
    #pragma unroll 1
    for (int it = 0; it < 100; ++it) {
      float e1 = 1.f - eg * __expf(-GAMMA * xm1);
      float f1 = e1 * e1 * xm1 - zperp;
      float e2 = 1.f - eg * __expf(-GAMMA * xm2);
      float f2 = e2 * e2 * xm2 - zperp;
      float fd = f1 - f2;
      if (fabsf(fd) < TOLV) fd = (fd >= 0.f) ? TOLV : -TOLV;
      float x = xm1 - f1 * (xm1 - xm2) / fd;
      xm2 = xm1; xm1 = x;
    }
    kern_s[r] = ALPHA * __expf(-GAMMA * (zun + xm1));
  }
  __syncthreads();

  // ---- phase 3: H = triu products of coord = ZU - x_cU, bf16, K-padded ----
  for (int idx = tid; idx < MTILE * FPAD; idx += 256) {
    int r = idx / FPAD, p = idx - r * FPAD;
    unsigned short h = 0;
    if (p < NFEAT) {
      int i = iu_s[p], j = ju_s[p];
      float a = ZUs[r * KDIM + i] - xcU_s[i];
      float b = ZUs[r * KDIM + j] - xcU_s[j];
      h = f2bf(a * b);
    }
    Hs[r * HS_STRIDE + p] = h;
  }
  __syncthreads();

  // ---- phase 4: WMMA  C(32x512) = H(32x544) @ Bpack(544x512) ----
  const int wave  = tid >> 5, lane = tid & 31;
  const int lhalf = lane >> 4, l16 = lane & 15;
  const int msub  = (wave >> 2) * 16;       // 0 or 16
  const int n0    = (wave & 3) * 128;       // column slab base

  v8f acc[8];
  #pragma unroll
  for (int t = 0; t < 8; ++t) acc[t] = (v8f){0.f,0.f,0.f,0.f,0.f,0.f,0.f,0.f};

  const unsigned short* arow = Hs + (msub + l16) * HS_STRIDE;

  for (int kb = 0; kb < KBLKS; ++kb) {
    const int kk = kb * 32;
    ABFrag A;   // A in contiguous k order, same as B: lanes0-15: K0..15; lanes16-31: K16..31
    A.h[0] = *(const u16x8*)(arow + kk + lhalf * 16);
    A.h[1] = *(const u16x8*)(arow + kk + lhalf * 16 + 8);
    #pragma unroll
    for (int t = 0; t < 8; ++t) {
      const int c = n0 + t * 16 + l16;
      const unsigned short* bptr = Bp + ((size_t)(kb * DDIM + c) * 2 + lhalf) * 16;
      ABFrag B;   // lanes0-15: K0..15 of column c; lanes16-31: K16..31
      B.h[0] = *(const u16x8*)(bptr);
      B.h[1] = *(const u16x8*)(bptr + 8);
      acc[t] = __builtin_amdgcn_wmma_f32_16x16x32_bf16(
          false, A.v, false, B.v, (short)0, acc[t], false, false);
    }
  }

  // ---- phase 5: fused epilogue  out = Z + kern*(change + C) ----
  #pragma unroll
  for (int t = 0; t < 8; ++t) {
    const int c = n0 + t * 16 + l16;
    const float chg = chg_s[c];
    #pragma unroll
    for (int v = 0; v < 8; ++v) {
      const int rloc = msub + v + lhalf * 8;   // C/D layout: VGPR v -> M=v (+8 hi lanes)
      float res = Zs[rloc * ZS_STRIDE + c] + kern_s[rloc] * (chg + acc[t][v]);
      __builtin_nontemporal_store(res, out + (size_t)(row0 + rloc) * DDIM + c);
    }
  }
}

extern "C" void kernel_launch(void* const* d_in, const int* in_sizes, int n_in,
                              void* d_out, int out_size, void* d_ws, size_t ws_size,
                              hipStream_t stream) {
  const float* Z   = (const float*)d_in[0];
  const float* U   = (const float*)d_in[1];
  const float* V   = (const float*)d_in[2];
  const float* xmu = (const float*)d_in[3];
  const float* xc  = (const float*)d_in[4];
  float* out = (float*)d_out;

  float* wsf = (float*)d_ws;
  unsigned short* Bp = (unsigned short*)((char*)d_ws + WS_BPACK_B);

  prep_proj<<<1, 64, 0, stream>>>(U, xmu, xc, wsf);
  prep_change<<<2, 256, 0, stream>>>(U, xmu, xc, wsf);
  const int nbp = KBLKS * DDIM * 32;
  prep_bpack<<<(nbp + 255) / 256, 256, 0, stream>>>(V, Bp);

  // dynamic LDS: Zs 66048 + Hs 35328 + ZUs 4096 + xmu 2048 + change 2048
  //            + xmuU/xcU/kern 384 + iu/ju 1088 + red 1024 = 112064 bytes
  const size_t smem = 66048 + 35328 + 4096 + 2048 + 2048 + 128 + 128 + 128 + 544 + 544 + 1024;
  glayer_main<<<N_ROWS / MTILE, 256, smem, stream>>>(Z, U, xmu, wsf, Bp, out);
}
